// LiftSplatShoot_90520730731185
// MI455X (gfx1250) — matrix operand order repaired and checked by reading, not compiled, not run on hardware
//
#include <hip/hip_runtime.h>
#include <hip/hip_bf16.h>

typedef __attribute__((ext_vector_type(16))) _Float16 v16h;
typedef __attribute__((ext_vector_type(8)))  _Float16 v8h;
typedef __attribute__((ext_vector_type(8)))  float    v8f;

// ---------------------------------------------------------------------------
// Implicit-GEMM convolution using CDNA5 WMMA (f16 in, f32 accum).
// K is tap-major: K = (r*KS+s)*cin + ci  (no integer division in hot loop).
//
// Pre-converted f16 activations (NCHW) + pre-packed f16 weights
// wpk[row][tap][ci] (rows zero-padded to the block grid):
//   A fragment = two contiguous 16B global loads per K-step,
//   B fragments = contiguous 32B LDS reads from a cooperatively staged tile.
// Block tile: 128 (Cout) x 64 (pixels); 8 waves share the B tile via LDS.
// Each wave: 1 A fragment -> 4 WMMAs per K-step. LDS tile is double-buffered
// so only ONE barrier per K-step is needed.
// ---------------------------------------------------------------------------
template <int KS>
__global__ void __launch_bounds__(256)
conv_wmma_kernel(const _Float16* __restrict__ in, const _Float16* __restrict__ wpk,
                 const float* __restrict__ bias, float* __restrict__ out,
                 int nimg, int cin, int cout, int hin, int win,
                 int hout, int wout, int stride, int pad)
{
    const int tid  = threadIdx.x;
    const int lane = tid & 31;
    const int wv   = tid >> 5;           // wave in block -> Cout sub-tile
    const int half = lane >> 4;
    const int l16  = lane & 15;

    const int P        = hout * wout;
    const int tilesN   = (P + 63) >> 6;
    const int tilesM   = (cout + 15) >> 4;
    const int tmGroups = (tilesM + 7) >> 3;

    // blockIdx.x = (img * tmGroups + tg) * tilesN + tn   -- every block active
    int bid = blockIdx.x;
    const int tn  = bid % tilesN;  bid /= tilesN;
    const int tg  = bid % tmGroups;
    const int img = bid / tmGroups;

    const int  tm  = tg * 8 + wv;
    const int  m   = tm * 16 + l16;        // padded weight row (always in wpk)
    constexpr int KSQ = KS * KS;
    const long hw  = (long)hin * win;

    const _Float16* inb = in + (long)img * cin * hw;

    // Cooperative-loader mapping: thread owns pixel column px (0..63) and
    // K rows kb0*8 .. kb0*8+7 of the 64x32 tile.
    const int  px   = tid & 63;
    const int  kb0  = tid >> 6;            // 0..3
    const int  n_ld = tn * 64 + px;
    const bool pvld = (n_ld < P);
    const int  nn   = pvld ? n_ld : 0;
    const int  oh_ld = nn / wout;
    const int  ow_ld = nn - oh_ld * wout;

    __shared__ __align__(32) _Float16 ldsB[2 * 64 * 32];   // double-buffered [px][k]

    v8f acc[4] = {v8f{}, v8f{}, v8f{}, v8f{}};
    int parity = 0;

#pragma unroll
    for (int tap = 0; tap < KSQ; ++tap) {
        const int r = tap / KS;
        const int s = tap - r * KS;

        // loader-side source for this tap (guarded by lv)
        const int  ih = oh_ld * stride + r - pad;
        const int  iw = ow_ld * stride + s - pad;
        const bool lv = pvld && ih >= 0 && ih < hin && iw >= 0 && iw < win;
        const _Float16* bsrc = inb + ((long)(lv ? ih : 0) * win + (lv ? iw : 0));
        // packed A row for this tap: contiguous along ci
        const _Float16* asrc = wpk + ((long)m * KSQ + tap) * cin;

        for (int ci0 = 0; ci0 < cin; ci0 += 32) {
            _Float16* buf = ldsB + parity * (64 * 32);
            parity ^= 1;
            __builtin_prefetch(asrc + ci0 + 256, 0, 1);
#pragma unroll
            for (int q = 0; q < 8; ++q) {
                const int k = kb0 * 8 + q;
                _Float16 v = (_Float16)0.0f;
                if (lv) v = bsrc[(long)(ci0 + k) * hw];
                buf[px * 32 + k] = v;
            }
            __syncthreads();             // producers done; other buffer free

            // A fragment: K = half*16 + e
            const v8h alo = *(const v8h*)(asrc + ci0 + half * 16);
            const v8h ahi = *(const v8h*)(asrc + ci0 + half * 16 + 8);
            v16h afrag;
#pragma unroll
            for (int e = 0; e < 8; ++e) {
                afrag[e]     = alo[e];
                afrag[8 + e] = ahi[e];
            }
            // B fragments: wave columns j*16 + l16, K = half*16 + e.
#pragma unroll
            for (int j = 0; j < 4; ++j) {
                const v16h bj = *(const v16h*)(buf + (j * 16 + l16) * 32 + half * 16);
                acc[j] = __builtin_amdgcn_wmma_f32_16x16x32_f16(
                    false, afrag, false, bj, (short)0, acc[j], false, false);
            }
        }
    }

    // C/D layout: VGPR r, lane -> M = tm*16 + r + 8*half, N = tn*64 + j*16 + l16
#pragma unroll
    for (int r = 0; r < 8; ++r) {
        const int om = tm * 16 + r + 8 * half;
        if (om >= cout) continue;
        const float bs    = bias ? bias[om] : 0.0f;
        const long  obase = ((long)img * cout + om) * P;
#pragma unroll
        for (int j = 0; j < 4; ++j) {
            const int n = tn * 64 + j * 16 + l16;
            if (n < P) out[obase + n] = acc[j][r] + bs;
        }
    }
}

// ---------------------------------------------------------------------------
// Weight repack: w (cout, cin, ks, ks) f32 -> wpk[row][tap][ci] f16,
// rows zero-padded up to rowsPad. blockIdx.y = row*ksq + tap.
// ---------------------------------------------------------------------------
__global__ void __launch_bounds__(256)
repack_w_kernel(const float* __restrict__ w, _Float16* __restrict__ wpk,
                int cout, int cin, int ksq)
{
    const int ci = blockIdx.x * blockDim.x + threadIdx.x;
    if (ci >= cin) return;
    const int row = blockIdx.y / ksq;
    const int tap = blockIdx.y - row * ksq;
    float v = 0.0f;
    if (row < cout) v = w[((long)row * cin + ci) * ksq + tap];
    wpk[(long)blockIdx.y * cin + ci] = (_Float16)v;
}

__global__ void __launch_bounds__(256)
f32_to_f16_kernel(const float* __restrict__ s, _Float16* __restrict__ d, long n)
{
    const long i = (long)blockIdx.x * blockDim.x + threadIdx.x;
    if (i < n) d[i] = (_Float16)s[i];
}

// ---------------------------------------------------------------------------
// Per-channel batch statistics: mean + rsqrt(var + eps) over (img, h, w).
// ---------------------------------------------------------------------------
__global__ void __launch_bounds__(256)
chan_stats_kernel(const float* __restrict__ x, float* __restrict__ mean,
                  float* __restrict__ rstd, int nimg, int C, int P, float eps)
{
    const int c = blockIdx.x;
    float s = 0.0f, s2 = 0.0f;
    for (int im = 0; im < nimg; ++im) {
        const float* base = x + ((long)im * C + c) * P;
        for (int p = threadIdx.x; p < P; p += blockDim.x) {
            const float v = base[p];
            s += v; s2 += v * v;
        }
    }
    __shared__ float sh1[256];
    __shared__ float sh2[256];
    sh1[threadIdx.x] = s;
    sh2[threadIdx.x] = s2;
    __syncthreads();
    for (int o = 128; o > 0; o >>= 1) {
        if ((int)threadIdx.x < o) {
            sh1[threadIdx.x] += sh1[threadIdx.x + o];
            sh2[threadIdx.x] += sh2[threadIdx.x + o];
        }
        __syncthreads();
    }
    if (threadIdx.x == 0) {
        const float n   = (float)((long)nimg * P);
        const float m   = sh1[0] / n;
        const float var = sh2[0] / n - m * m;
        mean[c] = m;
        rstd[c] = rsqrtf(var + eps);
    }
}

// ---------------------------------------------------------------------------
// BN (batch-stat) + ReLU. 2-D grid: y = (img*C + c), x covers pixels.
// ---------------------------------------------------------------------------
__global__ void __launch_bounds__(256)
bn_relu_f16_kernel(const float* __restrict__ x, _Float16* __restrict__ y,
                   const float* __restrict__ g, const float* __restrict__ b,
                   const float* __restrict__ mean, const float* __restrict__ rstd,
                   int C, int P)
{
    const int p = blockIdx.x * blockDim.x + threadIdx.x;
    if (p >= P) return;
    const int c = blockIdx.y % C;
    const long i = (long)blockIdx.y * P + p;
    const float v = g[c] * (x[i] - mean[c]) * rstd[c] + b[c];
    y[i] = (_Float16)(v > 0.0f ? v : 0.0f);
}

__global__ void __launch_bounds__(256)
bn_relu_f32_kernel(const float* __restrict__ x, float* __restrict__ y,
                   const float* __restrict__ g, const float* __restrict__ b,
                   const float* __restrict__ mean, const float* __restrict__ rstd,
                   int C, int P)
{
    const int p = blockIdx.x * blockDim.x + threadIdx.x;
    if (p >= P) return;
    const int c = blockIdx.y % C;
    const long i = (long)blockIdx.y * P + p;
    const float v = g[c] * (x[i] - mean[c]) * rstd[c] + b[c];
    y[i] = v > 0.0f ? v : 0.0f;
}

// ---------------------------------------------------------------------------
// Softmax over the first 118 channels (depth) + split of the 64 ctx channels.
// ---------------------------------------------------------------------------
__global__ void __launch_bounds__(256)
softmax_split_kernel(const float* __restrict__ x, float* __restrict__ depth,
                     float* __restrict__ ctx, int nimg, int P)
{
    const int i = blockIdx.x * blockDim.x + threadIdx.x;
    if (i >= nimg * P) return;
    const int img = i / P;
    const int p   = i - img * P;
    const float* col = x + (long)img * 182 * P + p;
    float mx = -3.0e38f;
    for (int d = 0; d < 118; ++d) mx = fmaxf(mx, col[(long)d * P]);
    float s = 0.0f;
    for (int d = 0; d < 118; ++d) s += __expf(col[(long)d * P] - mx);
    const float inv = 1.0f / s;
    for (int d = 0; d < 118; ++d)
        depth[((long)img * 118 + d) * P + p] = __expf(col[(long)d * P] - mx) * inv;
    for (int c = 0; c < 64; ++c)
        ctx[((long)img * 64 + c) * P + p] = col[(long)(118 + c) * P];
}

// ---------------------------------------------------------------------------
// Geometry + splat. One wave per lifted point (b,n,d,h,w); lane l handles
// context channels l and l+32. Truncating casts match .astype(int32).
// ---------------------------------------------------------------------------
__global__ void __launch_bounds__(256)
splat_kernel(const float* __restrict__ ctx, const float* __restrict__ depth,
             const float* __restrict__ intr, const float* __restrict__ extr,
             float* __restrict__ bev, long npts)
{
    const long t   = (long)blockIdx.x * blockDim.x + threadIdx.x;
    const long pt  = t >> 5;
    const int lane = (int)(t & 31);
    if (pt >= npts) return;

    const int p = (int)(pt % 704);
    long tmp    = pt / 704;
    const int d = (int)(tmp % 118); tmp /= 118;
    const int n = (int)(tmp % 6);
    const int b = (int)(tmp / 6);
    const int h = p / 44;
    const int w = p - h * 44;

    const float dep = 1.0f + 0.5f * (float)d;

    const float* K = intr + ((long)b * 6 + n) * 9;
    const float a00 = K[0], a01 = K[1], a02 = K[2];
    const float a10 = K[3], a11 = K[4], a12 = K[5];
    const float a20 = K[6], a21 = K[7], a22 = K[8];
    const float det = a00 * (a11 * a22 - a12 * a21)
                    - a01 * (a10 * a22 - a12 * a20)
                    + a02 * (a10 * a21 - a11 * a20);
    const float id = 1.0f / det;
    const float i00 = (a11 * a22 - a12 * a21) * id, i01 = (a02 * a21 - a01 * a22) * id, i02 = (a01 * a12 - a02 * a11) * id;
    const float i10 = (a12 * a20 - a10 * a22) * id, i11 = (a00 * a22 - a02 * a20) * id, i12 = (a02 * a10 - a00 * a12) * id;
    const float i20 = (a10 * a21 - a11 * a20) * id, i21 = (a01 * a20 - a00 * a21) * id, i22 = (a00 * a11 - a01 * a10) * id;

    const float px = (float)w * dep, py = (float)h * dep, pz = dep;
    const float cx = i00 * px + i01 * py + i02 * pz;
    const float cy = i10 * px + i11 * py + i12 * pz;
    const float cz = i20 * px + i21 * py + i22 * pz;

    const float* E = extr + ((long)b * 6 + n) * 16;
    const float ex = E[0] * cx + E[1] * cy + E[2]  * cz + E[3];
    const float ey = E[4] * cx + E[5] * cy + E[6]  * cz + E[7];
    const float ez = E[8] * cx + E[9] * cy + E[10] * cz + E[11];

    const int bx = (int)((ex + 50.0f) * 2.0f);
    const int by = (int)((ey + 50.0f) * 2.0f);
    if (bx < 0 || bx >= 200 || by < 0 || by >= 200 || !(ez > 0.0f)) return;

    const long img  = (long)b * 6 + n;
    const float dv  = depth[(img * 118 + d) * 704 + p];
    const long base = (long)b * 64 * 40000 + (long)by * 200 + bx;
#pragma unroll
    for (int cc = 0; cc < 2; ++cc) {
        const int c = lane + 32 * cc;
        const float v = ctx[(img * 64 + c) * 704 + p] * dv;
        atomicAdd(&bev[base + (long)c * 40000], v);
    }
}

__global__ void __launch_bounds__(256)
fill0_kernel(float* __restrict__ p, long n)
{
    const long i = (long)blockIdx.x * blockDim.x + threadIdx.x;
    if (i < n) p[i] = 0.0f;
}

// ---------------------------------------------------------------------------
// Host-side helpers
// ---------------------------------------------------------------------------
static inline void launch_repack(const float* w, _Float16* wpk, int cout, int cin,
                                 int ks, hipStream_t stream)
{
    const int ksq = ks * ks;
    const int rowsPad = ((((cout + 15) / 16) + 7) / 8) * 128;  // block grid coverage
    dim3 grid((cin + 255) / 256, rowsPad * ksq);
    repack_w_kernel<<<grid, 256, 0, stream>>>(w, wpk, cout, cin, ksq);
}

static inline void launch_conv(const _Float16* in, const _Float16* wpk,
                               const float* bias, float* out, int nimg, int cin,
                               int cout, int hin, int win, int ks, int stride,
                               int pad, hipStream_t stream)
{
    const int hout = (hin + 2 * pad - ks) / stride + 1;
    const int wout = (win + 2 * pad - ks) / stride + 1;
    const long P        = (long)hout * wout;
    const long tilesN   = (P + 63) / 64;
    const long tmGroups = (((cout + 15) / 16) + 7) / 8;
    const long blocks   = (long)nimg * tmGroups * tilesN;
    if (ks == 3) {
        conv_wmma_kernel<3><<<dim3((unsigned)blocks), 256, 0, stream>>>(
            in, wpk, bias, out, nimg, cin, cout, hin, win, hout, wout, stride, pad);
    } else {
        conv_wmma_kernel<1><<<dim3((unsigned)blocks), 256, 0, stream>>>(
            in, wpk, bias, out, nimg, cin, cout, hin, win, hout, wout, stride, pad);
    }
}

extern "C" void kernel_launch(void* const* d_in, const int* in_sizes, int n_in,
                              void* d_out, int out_size, void* d_ws, size_t ws_size,
                              hipStream_t stream)
{
    (void)in_sizes; (void)n_in; (void)out_size; (void)ws_size;

    const float* features = (const float*)d_in[0];   // (2,6,512,16,44)
    const float* intr     = (const float*)d_in[1];
    const float* extr     = (const float*)d_in[2];
    const float* dn_w1    = (const float*)d_in[3];
    const float* dn_b1    = (const float*)d_in[4];
    const float* dn_g1    = (const float*)d_in[5];
    const float* dn_be1   = (const float*)d_in[6];
    const float* dn_w2    = (const float*)d_in[7];
    const float* dn_b2    = (const float*)d_in[8];

    float* out = (float*)d_out;               // [bev 320000 | depth 996864]
    char*  wsb = (char*)d_ws;

    // ---- workspace byte layout (~76.5 MB) ----
    _Float16* actA = (_Float16*)(wsb);                       // 10,240,000 B
    _Float16* actB = (_Float16*)(wsb + 10240000);            // 20,480,000 B
    float*    rf32 = (float*)   (wsb + 30720000);            // 40,960,000 B
    float*    bev  = rf32;                                   // [0, 20.48MB) of rf32
    float*    ctxb = (float*)   (wsb + 30720000 + 32000000); // 2,162,688 B
    _Float16* wpk  = (_Float16*)(wsb + 71680000);            // 4,718,592 B
    float*    mean = (float*)   (wsb + 76399616);
    float*    rstd = mean + 512;

    // ---- DepthNet ----
    const long nfeat = 12L * 512 * 704;
    f32_to_f16_kernel<<<dim3((unsigned)((nfeat + 255) / 256)), 256, 0, stream>>>(
        features, actA, nfeat);

    launch_repack(dn_w1, wpk, 512, 512, 3, stream);
    launch_conv(actA, wpk, dn_b1, rf32, 12, 512, 512, 16, 44, 3, 1, 1, stream);
    chan_stats_kernel<<<512, 256, 0, stream>>>(rf32, mean, rstd, 12, 512, 704, 1e-5f);
    bn_relu_f16_kernel<<<dim3((704 + 255) / 256, 12 * 512), 256, 0, stream>>>(
        rf32, actB, dn_g1, dn_be1, mean, rstd, 512, 704);

    launch_repack(dn_w2, wpk, 182, 512, 1, stream);
    launch_conv(actB, wpk, dn_b2, rf32, 12, 512, 182, 16, 44, 1, 1, 0, stream);

    float* depth_out = out + 320000;          // (2,6,118,16,44)
    softmax_split_kernel<<<dim3((12 * 704 + 255) / 256), 256, 0, stream>>>(
        rf32, depth_out, ctxb, 12, 704);

    // ---- Splat (bev f32 lives at rf32 offset 0; c2 is dead by now) ----
    {
        const long nbev = 2L * 64 * 40000;
        fill0_kernel<<<dim3((unsigned)((nbev + 255) / 256)), 256, 0, stream>>>(bev, nbev);
        const long npts = 2L * 6 * 118 * 16 * 44;    // 996,864
        const long nthr = npts * 32;
        splat_kernel<<<dim3((unsigned)((nthr + 255) / 256)), 256, 0, stream>>>(
            ctxb, depth_out, intr, extr, bev, npts);
        f32_to_f16_kernel<<<dim3((unsigned)((nbev + 255) / 256)), 256, 0, stream>>>(
            bev, actA, nbev);                        // bevH -> actA
    }

    // ---- BEV encoder: repack -> conv(f16) -> stats -> BN+ReLU(f16) ----
    struct Layer { int wi, gi, bi, cin, cout, hin, win, stride; };
    const Layer layers[7] = {
        { 9, 10, 11,  64, 128, 200, 200, 1},  // e0
        {12, 13, 14, 128, 128, 200, 200, 2},  // e0s
        {15, 16, 17, 128, 256, 100, 100, 1},  // e1
        {18, 19, 20, 256, 256, 100, 100, 2},  // e1s
        {21, 22, 23, 256, 256,  50,  50, 1},  // e2
        {24, 25, 26, 256, 256,  50,  50, 2},  // e2s
        {27, 28, 29, 256, 256,  25,  25, 1},  // e3
    };
    for (int li = 0; li < 7; ++li) {
        const Layer& L = layers[li];
        const _Float16* src = (li & 1) ? actB : actA;   // e0 reads actA (bevH)
        _Float16* dstAct    = (li & 1) ? actA : actB;
        launch_repack((const float*)d_in[L.wi], wpk, L.cout, L.cin, 3, stream);
        launch_conv(src, wpk, nullptr, rf32, 2, L.cin, L.cout, L.hin, L.win,
                    3, L.stride, 1, stream);
        const int hout = (L.hin + 2 - 3) / L.stride + 1;
        const int wout = (L.win + 2 - 3) / L.stride + 1;
        const int P    = hout * wout;
        chan_stats_kernel<<<dim3((unsigned)L.cout), 256, 0, stream>>>(
            rf32, mean, rstd, 2, L.cout, P, 1e-5f);
        dim3 bgrid((P + 255) / 256, 2 * L.cout);
        if (li < 6) {
            bn_relu_f16_kernel<<<bgrid, 256, 0, stream>>>(
                rf32, dstAct, (const float*)d_in[L.gi], (const float*)d_in[L.bi],
                mean, rstd, L.cout, P);
        } else {
            // final layer: (2,256,25,25) f32 straight into d_out
            bn_relu_f32_kernel<<<bgrid, 256, 0, stream>>>(
                rf32, out, (const float*)d_in[L.gi], (const float*)d_in[L.bi],
                mean, rstd, L.cout, P);
        }
    }
}
